// Encoder_91276644975069
// MI455X (gfx1250) — matrix operand order repaired and checked by reading, not compiled, not run on hardware
//
#include <hip/hip_runtime.h>
#include <hip/hip_bf16.h>

// ---------------------------------------------------------------------------
// GAE encoder on MI455X (gfx1250, wave32).
//   - Dense GEMMs via v_wmma_f32_16x16x32_f16, K fully unrolled (2 / 4 WMMAs),
//     A pre-converted to f16 (b128 loads), B pre-packed into the per-lane
//     WMMA operand layout (b128 loads, wave reads 1KB contiguous).
//   - Edge aggregation: wave-per-edge coalesced gather + f32 global atomics
//     (the true bottleneck: ~3GB of L2-resident traffic; HBM never limits).
// ---------------------------------------------------------------------------

#define NNODES 100000
#define NEDGES 1600000
#define NLBL   100000
#define EMB_D  64
#define HID_D  128
#define OUT_D  64

typedef __attribute__((ext_vector_type(16))) _Float16 v16h;
typedef __attribute__((ext_vector_type(8)))  _Float16 v8h;
typedef __attribute__((ext_vector_type(8)))  float    v8f;

// ---------------- elementwise helpers ----------------

__global__ void k_fill(float* __restrict__ p, float v, int n) {
  int i = blockIdx.x * blockDim.x + threadIdx.x;
  if (i < n) p[i] = v;
}

__global__ void k_deg_accum(const int* __restrict__ dst, float* __restrict__ deg, int nE) {
  int i = blockIdx.x * blockDim.x + threadIdx.x;
  if (i < nE) atomicAdd(&deg[dst[i]], 1.0f);
}

__global__ void k_rsqrt_inplace(float* __restrict__ p, int n) {
  int i = blockIdx.x * blockDim.x + threadIdx.x;
  if (i < n) p[i] = rsqrtf(p[i]);
}

// out[node,f] = pre[node,f] * dis[node]^2 + bias[f]   (self-loop term + bias)
__global__ void k_agg_init(const float* __restrict__ pre, const float* __restrict__ dis,
                           const float* __restrict__ bias, float* __restrict__ out,
                           int total, int F) {
  int i = blockIdx.x * blockDim.x + threadIdx.x;
  if (i >= total) return;
  int node = i / F, f = i - node * F;
  float d = dis[node];
  out[i] = pre[i] * d * d + bias[f];
}

__global__ void k_relu(float* __restrict__ p, int n) {
  int i = blockIdx.x * blockDim.x + threadIdx.x;
  if (i < n) p[i] = fmaxf(p[i], 0.0f);
}

__global__ void k_clampmax(float* __restrict__ p, int n, float mx) {
  int i = blockIdx.x * blockDim.x + threadIdx.x;
  if (i < n) p[i] = fminf(p[i], mx);
}

__global__ void k_zero2(float* __restrict__ acc) {
  if (threadIdx.x < 2 && blockIdx.x == 0) acc[threadIdx.x] = 0.0f;
}

// f32 -> f16 mirror (compiler packs to v_cvt_pk_f16_f32 on b32 stores)
__global__ void k_cvt_f16(const float* __restrict__ src, _Float16* __restrict__ dst, int n) {
  int i = (blockIdx.x * blockDim.x + threadIdx.x) * 2;
  if (i + 1 < n) {
    dst[i]     = (_Float16)src[i];
    dst[i + 1] = (_Float16)src[i + 1];
  }
}

// ---------------- B operand pre-pack ----------------
// Packs f32 row-major B[K,Nn] into a contiguous-k (k = 16*hl + e) wave32 B layout matching A so the GEMM can
// load each lane's 16 halfs as two b128s:
//   Bp[((tn*ksteps + ks)*32 + lane)*16 + 2v+s] = B[ks*32 + 16*hl + 2v + s][tn*16 + ml]
template <int K, int Nn>
__global__ void k_packB(const float* __restrict__ B, _Float16* __restrict__ Bp) {
  constexpr int ksteps = K / 32;
  constexpr int ntn = Nn / 16;
  int t = blockIdx.x * blockDim.x + threadIdx.x;
  if (t >= ntn * ksteps * 32) return;
  int lane = t & 31;
  int ks   = (t >> 5) % ksteps;
  int tn   = (t >> 5) / ksteps;
  int hl = lane >> 4, ml = lane & 15;
  int col = tn * 16 + ml;
  _Float16* __restrict__ dst = Bp + (size_t)t * 16;
#pragma unroll
  for (int v = 0; v < 8; ++v) {
    int kr = ks * 32 + 16 * hl + 2 * v;
    dst[2 * v]     = (_Float16)B[(size_t)kr * Nn + col];
    dst[2 * v + 1] = (_Float16)B[(size_t)(kr + 1) * Nn + col];
  }
}

// ---------------- WMMA GEMM: C[M,Nn] = Ah[M,K](f16) * Bp(packed f16) -------
// One wave per 16x16 C tile, K fully unrolled. A/B: contiguous k = 16*hl + e; C layout per
// CDNA5 ISA 7.12.2 (wave32).
template <int K, int Nn>
__global__ __launch_bounds__(256)
void k_gemm_wmma(const _Float16* __restrict__ Ah, const _Float16* __restrict__ Bp,
                 float* __restrict__ C, int M) {
  constexpr int ksteps = K / 32;
  constexpr int ntn = Nn / 16;
  int wave = (blockIdx.x * blockDim.x + threadIdx.x) >> 5;
  int lane = threadIdx.x & 31;
  int tm = wave / ntn;
  int tn = wave - tm * ntn;
  if (tm * 16 >= M) return;               // wave-uniform: EXEC stays all-ones
  int m0 = tm << 4, n0 = tn << 4;
  int hl = lane >> 4;                     // lane half (0/1)
  int ml = lane & 15;
  // A: elements 0..15 = K = k0 + 16*hl .. +15
  const _Float16* __restrict__ arow = Ah + (size_t)(m0 + ml) * K + 16 * hl;
  const _Float16* __restrict__ blan = Bp + ((size_t)(tn * ksteps) * 32 + lane) * 16;

  v8f acc = {};
#pragma unroll
  for (int ks = 0; ks < ksteps; ++ks) {
    union { v16h v; v8h h[2]; } ua, ub;
    ua.h[0] = *(const v8h*)(arow + ks * 32);
    ua.h[1] = *(const v8h*)(arow + ks * 32 + 8);
    ub.h[0] = *(const v8h*)(blan + (size_t)ks * 32 * 16);
    ub.h[1] = *(const v8h*)(blan + (size_t)ks * 32 * 16 + 8);
    acc = __builtin_amdgcn_wmma_f32_16x16x32_f16(
        /*neg_a=*/false, ua.v, /*neg_b=*/false, ub.v,
        /*c_mod=*/(short)0, acc, /*reuse_a=*/false, /*reuse_b=*/false);
  }
  // C 16x16 f32 layout: VGPR r -> row m0 + r + 8*hl, col n0 + ml
#pragma unroll
  for (int r = 0; r < 8; ++r)
    C[(size_t)(m0 + r + 8 * hl) * Nn + n0 + ml] = acc[r];
}

// ---------------- edge scatter: out[dst] += pre[src] * dis[src]*dis[dst] ----
// One wave per edge; lanes stride features -> coalesced loads + f32 atomics.
template <int F>
__global__ __launch_bounds__(256)
void k_scatter(const int* __restrict__ src, const int* __restrict__ dst,
               const float* __restrict__ dis, const float* __restrict__ pre,
               float* __restrict__ out, int nE) {
  int wave = (blockIdx.x * blockDim.x + threadIdx.x) >> 5;
  int lane = threadIdx.x & 31;
  if (wave >= nE) return;
  int s = src[wave], d = dst[wave];
  float norm = dis[s] * dis[d];
  const float* __restrict__ prow = pre + (size_t)s * F + lane;
  float* __restrict__ orow = out + (size_t)d * F + lane;
#pragma unroll
  for (int c = 0; c < F / 32; ++c)
    atomicAdd(&orow[32 * c], prow[32 * c] * norm);
}

// ---------------- decoder loss: wave-per-edge dot(z[s],z[d]) -> BCE sum -----

__global__ __launch_bounds__(256)
void k_loss_edges(const float* __restrict__ z, const int* __restrict__ sidx,
                  const int* __restrict__ didx, int nP, int neg_mode,
                  float* __restrict__ acc) {
  int wave = (blockIdx.x * blockDim.x + threadIdx.x) >> 5;
  int lane = threadIdx.x & 31;
  if (wave >= nP) return;
  const float* __restrict__ zs = z + (size_t)sidx[wave] * OUT_D;
  const float* __restrict__ zd = z + (size_t)didx[wave] * OUT_D;
  float p = zs[lane] * zd[lane] + zs[lane + 32] * zd[lane + 32];
#pragma unroll
  for (int o = 16; o > 0; o >>= 1) p += __shfl_down(p, o, 32);
  if (lane == 0) {
    float sig = 1.0f / (1.0f + expf(-p));
    float v = neg_mode ? logf(1.0f - sig + 1e-15f) : logf(sig + 1e-15f);
    atomicAdd(acc, v);
  }
}

__global__ void k_finalize(const float* __restrict__ acc, float* __restrict__ loss, float invP) {
  if (threadIdx.x == 0 && blockIdx.x == 0)
    loss[0] = -(acc[0] * invP) - (acc[1] * invP);
}

// ---------------------------------------------------------------------------

extern "C" void kernel_launch(void* const* d_in, const int* in_sizes, int n_in,
                              void* d_out, int out_size, void* d_ws, size_t ws_size,
                              hipStream_t stream) {
  (void)in_sizes; (void)n_in; (void)out_size; (void)ws_size;
  const float* emb = (const float*)d_in[0];
  const int*   ei  = (const int*)d_in[1];     // [2,E] flat: src row then dst row
  const float* W1  = (const float*)d_in[2];
  const float* b1  = (const float*)d_in[3];
  const float* Wmu = (const float*)d_in[4];
  const float* bmu = (const float*)d_in[5];
  const float* Wls = (const float*)d_in[6];
  const float* bls = (const float*)d_in[7];
  const int*   pos = (const int*)d_in[8];     // [2,P]
  const int*   neg = (const int*)d_in[9];     // [2,P]

  float* out    = (float*)d_out;
  float* z      = out;                                   // [N,64]
  float* logstd = out + (size_t)NNODES * OUT_D;          // [N,64]
  float* loss   = out + (size_t)2 * NNODES * OUT_D;      // [1]

  // workspace layout (~128.5 MB peak):
  char* ws = (char*)d_ws;
  float*    dis = (float*)ws;                                          // N f32
  float*    acc = (float*)(ws + (size_t)NNODES * 4);                   // 2 f32
  float*    hW  = (float*)(ws + 409600);                               // N*HID f32
  float*    h   = (float*)(ws + 409600 + (size_t)NNODES * HID_D * 4);  // N*HID f32
  _Float16* f16buf = (_Float16*)(ws + 409600 + (size_t)2 * NNODES * HID_D * 4); // N*HID f16
  char*     bp_base = ws + 409600 + (size_t)2 * NNODES * HID_D * 4
                         + (size_t)NNODES * HID_D * 2;
  _Float16* bp1  = (_Float16*)bp_base;                 // 8*2*32*16  halfs (16KB)
  _Float16* bpmu = (_Float16*)(bp_base + 16384);       // 4*4*32*16  halfs (16KB)
  _Float16* bpls = (_Float16*)(bp_base + 32768);       // 4*4*32*16  halfs (16KB)
  float* mup = hW;                                     // N*OUT (reuses dead hW)
  float* lsp = hW + (size_t)NNODES * OUT_D;            // N*OUT

  const int* esrc = ei;
  const int* edst = ei + NEDGES;

  // ---- degree / normalization -------------------------------------------
  k_fill<<<(NNODES + 255) / 256, 256, 0, stream>>>(dis, 1.0f, NNODES);
  k_deg_accum<<<(NEDGES + 255) / 256, 256, 0, stream>>>(edst, dis, NEDGES);
  k_rsqrt_inplace<<<(NNODES + 255) / 256, 256, 0, stream>>>(dis, NNODES);
  k_zero2<<<1, 32, 0, stream>>>(acc);

  // ---- layer 1: h = relu( D^-1/2 (A+I) D^-1/2 (emb @ W1) + b1 ) ---------
  k_cvt_f16<<<(NNODES * EMB_D / 2 + 255) / 256, 256, 0, stream>>>(emb, f16buf,
                                                                  NNODES * EMB_D);
  k_packB<EMB_D, HID_D><<<2, 256, 0, stream>>>(W1, bp1);
  // tiles = (N/16)*(HID/16) = 50000 waves; 8 waves/block
  k_gemm_wmma<EMB_D, HID_D><<<6250, 256, 0, stream>>>(f16buf, bp1, hW, NNODES);
  k_agg_init<<<(NNODES * HID_D + 255) / 256, 256, 0, stream>>>(hW, dis, b1, h,
                                                               NNODES * HID_D, HID_D);
  k_scatter<HID_D><<<(NEDGES + 7) / 8, 256, 0, stream>>>(esrc, edst, dis, hW, h, NEDGES);
  k_relu<<<(NNODES * HID_D + 255) / 256, 256, 0, stream>>>(h, NNODES * HID_D);

  // ---- layer 2: mu (=z) and logstd --------------------------------------
  k_cvt_f16<<<(NNODES * HID_D / 2 + 255) / 256, 256, 0, stream>>>(h, f16buf,
                                                                  NNODES * HID_D);
  k_packB<HID_D, OUT_D><<<2, 256, 0, stream>>>(Wmu, bpmu);
  k_packB<HID_D, OUT_D><<<2, 256, 0, stream>>>(Wls, bpls);
  // tiles = (N/16)*(OUT/16) = 25000 waves
  k_gemm_wmma<HID_D, OUT_D><<<3125, 256, 0, stream>>>(f16buf, bpmu, mup, NNODES);
  k_gemm_wmma<HID_D, OUT_D><<<3125, 256, 0, stream>>>(f16buf, bpls, lsp, NNODES);
  k_agg_init<<<(NNODES * OUT_D + 255) / 256, 256, 0, stream>>>(mup, dis, bmu, z,
                                                               NNODES * OUT_D, OUT_D);
  k_agg_init<<<(NNODES * OUT_D + 255) / 256, 256, 0, stream>>>(lsp, dis, bls, logstd,
                                                               NNODES * OUT_D, OUT_D);
  k_scatter<OUT_D><<<(NEDGES + 7) / 8, 256, 0, stream>>>(esrc, edst, dis, mup, z, NEDGES);
  k_scatter<OUT_D><<<(NEDGES + 7) / 8, 256, 0, stream>>>(esrc, edst, dis, lsp, logstd, NEDGES);
  k_clampmax<<<(NNODES * OUT_D + 255) / 256, 256, 0, stream>>>(logstd, NNODES * OUT_D, 10.0f);

  // ---- decoder BCE loss --------------------------------------------------
  k_loss_edges<<<(NLBL + 7) / 8, 256, 0, stream>>>(z, pos, pos + NLBL, NLBL, 0, acc + 0);
  k_loss_edges<<<(NLBL + 7) / 8, 256, 0, stream>>>(z, neg, neg + NLBL, NLBL, 1, acc + 1);
  k_finalize<<<1, 1, 0, stream>>>(acc, loss, 1.0f / (float)NLBL);
}
